// TreeFilter2D_11982958756212
// MI455X (gfx1250) — matrix operand order repaired and checked by reading, not compiled, not run on hardware
//
#include <hip/hip_runtime.h>

// ---------------------------------------------------------------------------
// TreeFilter2D for MI455X (gfx1250, wave32).
//
// B=4, C=128, CE=64, H=W=64 -> V=4096, MAX_LEVELS=14.
//
//  * Kernel 1: edge weights w[b,v] = exp(-||e_s[v]-e_s[par(v)]||^2).
//      d2 for a 16-column tile == diag(D^T D), D = 64x16 f16 diff tile
//      -> two V_WMMA_F32_16X16X32_F16 (K=64 as 2x K=32).
//      Diff tile staged column-major in LDS ([16 cols][72-half stride]) so
//      every WMMA fragment half is a contiguous 16B run -> ds_load_b128,
//      bank-conflict-free (base dword 36*j mod 64 distinct for j=0..15).
//      Diagonal extracted branchlessly from the C layout (VGPR i: lanes 0-15
//      hold (M=i,N=lane), lanes 16-31 hold (M=8+i,N=lane-16)).
//  * Kernel 2: per (batch, 8-channel chunk) workgroup keeps ALL sweep state
//      in 197KB LDS (CDNA5 WGP has 320KB): parent, w, nrm, level-bucketed
//      order, 8x4096 f32 aggr tile. Up-sweep: ds_add_f32 atomics (siblings
//      share parents). Down-sweep: in place (a node's pre-value is read only
//      by itself; parents finalized one level earlier). 64 independent WGs.
// ---------------------------------------------------------------------------

#define B_      4
#define C_      128
#define CE_     64
#define V_      4096
#define MAXLVL  14
#define CT      8                   // channels per tree workgroup
#define CSTRIDE 72                  // padded column stride (halves)

typedef __attribute__((ext_vector_type(16))) _Float16 v16h;
typedef __attribute__((ext_vector_type(8)))  _Float16 v8h;
typedef __attribute__((ext_vector_type(8)))  float    v8f;

// ---------------------------------------------------------------------------
// Kernel 1: WMMA edge weights.  grid = B * (V/64), 128 threads
// (4 waves; each wave owns one 16-column tile).
// ---------------------------------------------------------------------------
__global__ __launch_bounds__(128) void tf_weights_wmma(
    const float* __restrict__ embed,   // [B][CE][V]
    const int*   __restrict__ sidx,    // [B][V]
    const int*   __restrict__ spar,    // [B][V]
    float*       __restrict__ w_out)   // [B][V]
{
    // Column-major diff tile per wave: column j's 64 channels contiguous.
    __shared__ __align__(16) _Float16 ldsDT[4][16][CSTRIDE];

    const int tid  = threadIdx.x;
    const int wave = tid >> 5;
    const int lane = tid & 31;
    const int blocksPerBatch = V_ / 64;             // 64
    const int b  = blockIdx.x / blocksPerBatch;
    const int v0 = (blockIdx.x % blocksPerBatch) * 64 + wave * 16;

    const int j = lane & 15;                        // column within tile
    const int v = v0 + j;
    const int idx  = sidx[b * V_ + v];
    const int pidx = sidx[b * V_ + spar[b * V_ + v]];
    const float* eb = embed + (size_t)b * CE_ * V_;

    // Half-wave splits channels: lanes 0-15 do c=0..31, lanes 16-31 do 32..63.
    const int cb = (lane >> 4) * 32;
    #pragma unroll 4
    for (int c = cb; c < cb + 32; ++c) {
        float d = eb[c * V_ + idx] - eb[c * V_ + pidx];
        ldsDT[wave][j][c] = (_Float16)d;
    }
    __syncthreads();

    // Fragment packing (ISA 7.12.2, f16 16x16x32). In column-major LDS every
    // needed K-run is contiguous:
    //  A[m][k]=D[k][m]: lanes 0-15 (m=lane)  halves K={0..7},{16..23}
    //                   lanes16-31 (m=l-16)  halves K={8..15},{24..31}
    //  B[k][n]=D[k][n]: lanes 0-15 (n=lane)  halves K={0..7},{16..23}
    //                   lanes16-31 (n=l-16)  halves K={8..15},{24..31}
    const _Float16* col = &ldsDT[wave][j][0];
    const int ha = (lane < 16) ? 0 : 8;    // A first-run base
    const int hb = (lane < 16) ? 0 : 8;    // B first-run base

    v8h aL0 = *(const v8h*)(col + ha);
    v8h aH0 = *(const v8h*)(col + ha + 16);
    v8h bL0 = *(const v8h*)(col + hb);
    v8h bH0 = *(const v8h*)(col + hb + 16);
    v8h aL1 = *(const v8h*)(col + 32 + ha);
    v8h aH1 = *(const v8h*)(col + 32 + ha + 16);
    v8h bL1 = *(const v8h*)(col + 32 + hb);
    v8h bH1 = *(const v8h*)(col + 32 + hb + 16);

    v16h a0 = __builtin_shufflevector(aL0, aH0, 0,1,2,3,4,5,6,7,8,9,10,11,12,13,14,15);
    v16h b0 = __builtin_shufflevector(bL0, bH0, 0,1,2,3,4,5,6,7,8,9,10,11,12,13,14,15);
    v16h a1 = __builtin_shufflevector(aL1, aH1, 0,1,2,3,4,5,6,7,8,9,10,11,12,13,14,15);
    v16h b1 = __builtin_shufflevector(bL1, bH1, 0,1,2,3,4,5,6,7,8,9,10,11,12,13,14,15);

    v8f acc = {};
    acc = __builtin_amdgcn_wmma_f32_16x16x32_f16(false, a0, false, b0,
                                                 (short)0, acc, false, false);
    acc = __builtin_amdgcn_wmma_f32_16x16x32_f16(false, a1, false, b1,
                                                 (short)0, acc, false, false);

    // Branchless diag extract: diag(j,j) lives at (VGPR j, lane j) for j<8
    // and (VGPR j-8, lane j+16) for j>=8.  So lanes 0-7 and 24-31 each hold
    // exactly one diagonal element in acc[lane&7].
    const int  reg    = lane & 7;
    const bool active = (lane < 8) || (lane >= 24);
    const int  dstj   = (lane < 8) ? lane : (lane - 16);

    float d2 = acc[0];
    #pragma unroll
    for (int i = 1; i < 8; ++i) d2 = (reg == i) ? acc[i] : d2;

    if (active) w_out[(size_t)b * V_ + v0 + dstj] = __expf(-d2);
}

// ---------------------------------------------------------------------------
// Kernel 2: tree sweeps, all state in LDS.
// grid = B * (C/CT) = 64 blocks, 256 threads (8 waves).
// Sweep thread layout: c = tid&7 (channel), slot = tid>>3 (node slot).
// ---------------------------------------------------------------------------
#define SMEM_BYTES (V_ * 16 + CT * V_ * 4 + 48 * 4)   // 196,800 B

__global__ __launch_bounds__(256) void tf_tree(
    const float* __restrict__ feat,   // [B][C][V]
    const int*   __restrict__ sidx,   // [B][V]
    const int*   __restrict__ spar,   // [B][V]
    const int*   __restrict__ slvl,   // [B][V]
    const float* __restrict__ w_in,   // [B][V]
    float*       __restrict__ out)    // [B][C][V]
{
    extern __shared__ char smem[];
    int*   s_parent = (int*)  (smem);                         // V ints
    float* s_w      = (float*)(smem + V_ * 4);                // V floats
    float* s_nrm    = (float*)(smem + V_ * 8);                // V floats
    int*   s_order  = (int*)  (smem + V_ * 12);               // V ints
    float* s_aggr   = (float*)(smem + V_ * 16);               // CT*V floats
    int*   s_cnt    = (int*)  (smem + V_ * 16 + CT * V_ * 4); // 16
    int*   s_start  = s_cnt + 16;
    int*   s_off    = s_cnt + 32;

    const int tid   = threadIdx.x;
    const int gpb   = C_ / CT;                 // 16 channel-chunks per batch
    const int b     = blockIdx.x / gpb;
    const int cbase = (blockIdx.x % gpb) * CT;
    const int base  = b * V_;

    if (tid < 16) s_cnt[tid] = 0;
    __syncthreads();

    // Stage tree data + count level occupancy.
    for (int vv = tid; vv < V_; vv += 256) {
        s_parent[vv] = spar[base + vv];
        s_w[vv]      = w_in[base + vv];
        s_nrm[vv]    = 1.0f;
        int d = slvl[base + vv];
        if (d >= 1 && d < MAXLVL) atomicAdd(&s_cnt[d], 1);
    }
    __syncthreads();
    if (tid == 0) {
        int run = 0;
        for (int d = 1; d < MAXLVL; ++d) { s_start[d] = run; run += s_cnt[d]; }
    }
    __syncthreads();
    if (tid >= 1 && tid < MAXLVL) s_off[tid] = s_start[tid];
    __syncthreads();
    for (int vv = tid; vv < V_; vv += 256) {
        int d = slvl[base + vv];
        if (d >= 1 && d < MAXLVL) {
            int pos = atomicAdd(&s_off[d], 1);
            s_order[pos] = vv;
        }
    }
    // Gather feature chunk into LDS (aggr := f_s).
    for (int i = tid; i < CT * V_; i += 256) {
        int c  = i >> 12;            // /V_
        int vv = i & (V_ - 1);
        s_aggr[c * V_ + vv] =
            feat[((size_t)b * C_ + cbase + c) * V_ + sidx[base + vv]];
    }
    __syncthreads();

    const int c    = tid & (CT - 1);
    const int slot = tid >> 3;       // 0..31

    // Upward sweep (leaves -> root): level-d nodes push w*value into their
    // parents (level d-1). Siblings may share a parent -> ds_add_f32 atomics.
    for (int d = MAXLVL - 1; d >= 1; --d) {
        int beg = s_start[d], num = s_cnt[d];
        for (int s = slot; s < num; s += 32) {
            int vv   = s_order[beg + s];
            int p    = s_parent[vv];
            float wv = s_w[vv];
            atomicAdd(&s_aggr[c * V_ + p], wv * s_aggr[c * V_ + vv]);
            if (c == 0) atomicAdd(&s_nrm[p], wv * s_nrm[vv]);
        }
        __syncthreads();
    }

    // Downward sweep (root -> leaves), in place:
    //   a[v] = a_up[v] + w*(a[parent] - w*a_up[v]).
    for (int d = 1; d < MAXLVL; ++d) {
        int beg = s_start[d], num = s_cnt[d];
        for (int s = slot; s < num; s += 32) {
            int vv   = s_order[beg + s];
            int p    = s_parent[vv];
            float wv = s_w[vv];
            float au = s_aggr[c * V_ + vv];
            s_aggr[c * V_ + vv] = au + wv * (s_aggr[c * V_ + p] - wv * au);
            if (c == 0) {
                float nu = s_nrm[vv];
                s_nrm[vv] = nu + wv * (s_nrm[p] - wv * nu);
            }
        }
        __syncthreads();
    }

    // Normalize and scatter back through sorted_index (a permutation, so
    // every output element is written).
    for (int i = tid; i < CT * V_; i += 256) {
        int cc = i >> 12;
        int vv = i & (V_ - 1);
        out[((size_t)b * C_ + cbase + cc) * V_ + sidx[base + vv]] =
            s_aggr[cc * V_ + vv] / s_nrm[vv];
    }
}

// ---------------------------------------------------------------------------
extern "C" void kernel_launch(void* const* d_in, const int* in_sizes, int n_in,
                              void* d_out, int out_size, void* d_ws, size_t ws_size,
                              hipStream_t stream) {
    const float* feat = (const float*)d_in[0];   // feature_in   [B,C,H,W]
    const float* emb  = (const float*)d_in[1];   // embed_in     [B,CE,H,W]
    const int*   sidx = (const int*)  d_in[2];   // sorted_index [B,V]
    const int*   spar = (const int*)  d_in[3];   // sorted_parent[B,V]
    const int*   slvl = (const int*)  d_in[4];   // node_level   [B,V]
    float* out  = (float*)d_out;
    float* w_ws = (float*)d_ws;                  // B*V floats of scratch

    (void)in_sizes; (void)n_in; (void)out_size; (void)ws_size;

    // Allow the 197KB dynamic-LDS allocation (WGP has 320KB on CDNA5).
    hipFuncSetAttribute((const void*)tf_tree,
                        hipFuncAttributeMaxDynamicSharedMemorySize, SMEM_BYTES);

    // 1) edge weights via WMMA: B*(V/64) = 256 blocks x 128 threads.
    tf_weights_wmma<<<B_ * (V_ / 64), 128, 0, stream>>>(emb, sidx, spar, w_ws);

    // 2) tree sweeps: B*(C/CT) = 64 blocks x 256 threads, 197KB LDS each.
    tf_tree<<<B_ * (C_ / CT), 256, SMEM_BYTES, stream>>>(feat, sidx, spar,
                                                         slvl, w_ws, out);
}
